// SupervisedContrastiveLoss_16673063043356
// MI455X (gfx1250) — matrix operand order repaired and checked by reading, not compiled, not run on hardware
//
#include <hip/hip_runtime.h>
#include <hip/hip_bf16.h>
#include <math.h>

#define NROWS 8192
#define DDIM  1024
#define INV_T 10.0f   // 1 / 0.1

typedef __bf16 bf16_t;
typedef bf16_t v8bf  __attribute__((ext_vector_type(8)));
typedef bf16_t v16bf __attribute__((ext_vector_type(16)));
typedef float  v8f   __attribute__((ext_vector_type(8)));

static __device__ __forceinline__ unsigned short f32_to_bf16_rne(float f) {
    unsigned int u = __float_as_uint(f);
    unsigned int lsb = (u >> 16) & 1u;
    u += 0x7fffu + lsb;                 // round-to-nearest-even
    return (unsigned short)(u >> 16);
}
static __device__ __forceinline__ float bf16_to_f32(unsigned short h) {
    return __uint_as_float(((unsigned int)h) << 16);
}

// Reduce over the 16 lanes of each half-wave with ds_swizzle (group-of-32,
// and_mask=0x1f, xor_mask = 1,2,4,8).  One DS op per step, no lane-mask VALU.
static __device__ __forceinline__ float halfwave16_sum(float v) {
    v += __uint_as_float((unsigned)__builtin_amdgcn_ds_swizzle(
             (int)__float_as_uint(v), 0x041f));
    v += __uint_as_float((unsigned)__builtin_amdgcn_ds_swizzle(
             (int)__float_as_uint(v), 0x081f));
    v += __uint_as_float((unsigned)__builtin_amdgcn_ds_swizzle(
             (int)__float_as_uint(v), 0x101f));
    v += __uint_as_float((unsigned)__builtin_amdgcn_ds_swizzle(
             (int)__float_as_uint(v), 0x201f));
    return v;
}

// ---------------------------------------------------------------------------
// Kernel 1: row L2-normalize (fp32 in) -> bf16 out.  One block per row.
// ---------------------------------------------------------------------------
__global__ void k_normalize_bf16(const float* __restrict__ z,
                                 unsigned short* __restrict__ outbf) {
    const int row  = blockIdx.x;
    const int lane = threadIdx.x & 31;
    const int wave = threadIdx.x >> 5;
    const float* p = z + (size_t)row * DDIM;

    float ss = 0.0f;
    for (int i = threadIdx.x; i < DDIM; i += blockDim.x) {
        float v = p[i];
        ss += v * v;
    }
    #pragma unroll
    for (int m = 1; m < 32; m <<= 1) ss += __shfl_xor(ss, m, 32);

    __shared__ float red[8];
    if (lane == 0) red[wave] = ss;
    __syncthreads();
    if (wave == 0) {
        float v = (lane < (int)(blockDim.x >> 5)) ? red[lane] : 0.0f;
        #pragma unroll
        for (int m = 1; m < 8; m <<= 1) v += __shfl_xor(v, m, 32);
        if (lane == 0) red[0] = v;
    }
    __syncthreads();

    const float scale = 1.0f / fmaxf(sqrtf(red[0]), 1e-8f);
    unsigned short* o = outbf + (size_t)row * DDIM;
    for (int i = threadIdx.x; i < DDIM; i += blockDim.x)
        o[i] = f32_to_bf16_rne(p[i] * scale);
}

// ---------------------------------------------------------------------------
// Kernel 2: zero the denominator accumulators.
// ---------------------------------------------------------------------------
__global__ void k_zero_f32(float* __restrict__ p, int n) {
    int i = blockIdx.x * blockDim.x + threadIdx.x;
    if (i < n) p[i] = 0.0f;
}

// ---------------------------------------------------------------------------
// Kernel 3: diagonal similarities: diag[i] = dot(n1[i], n2[i]) / T.
// One wave per row (8 rows per 256-thread block).
// ---------------------------------------------------------------------------
__global__ void k_diag_sim(const unsigned short* __restrict__ n1,
                           const unsigned short* __restrict__ n2,
                           float* __restrict__ diag) {
    const int lane = threadIdx.x & 31;
    const int wave = threadIdx.x >> 5;
    const int row  = blockIdx.x * 8 + wave;

    const unsigned short* a = n1 + (size_t)row * DDIM;
    const unsigned short* b = n2 + (size_t)row * DDIM;

    float acc = 0.0f;
    #pragma unroll
    for (int k = lane * 4; k < DDIM; k += 32 * 4) {
        ushort4 ua = *(const ushort4*)(a + k);
        ushort4 ub = *(const ushort4*)(b + k);
        acc += bf16_to_f32(ua.x) * bf16_to_f32(ub.x);
        acc += bf16_to_f32(ua.y) * bf16_to_f32(ub.y);
        acc += bf16_to_f32(ua.z) * bf16_to_f32(ub.z);
        acc += bf16_to_f32(ua.w) * bf16_to_f32(ub.w);
    }
    #pragma unroll
    for (int m = 1; m < 32; m <<= 1) acc += __shfl_xor(acc, m, 32);
    if (lane == 0) diag[row] = acc * INV_T;
}

// ---------------------------------------------------------------------------
// Kernel 4: WMMA bf16 GEMM, 32x64 register tile per wave (2 row-frags x
// 4 col-frags = 8 WMMAs per K-step of 32), fused exp + row-sum epilogue.
//   block = 256 (8 waves, arranged 4 row-groups x 2 col-groups)
//   WG tile = 128 rows x 128 cols; grid = (64, 64).
// Loads per K-step: 2x2 (A) + 4x2 (B) = 12 b128 for 8 WMMAs (1.5 loads/wmma)
// vs 4 loads/wmma previously; WGP$ sees 128x128 tile reuse across waves.
// ---------------------------------------------------------------------------
__global__ void k_gemm_exp_rowsum(const unsigned short* __restrict__ n1,
                                  const unsigned short* __restrict__ n2,
                                  float* __restrict__ denom) {
    const int lane = threadIdx.x & 31;
    const int wave = threadIdx.x >> 5;          // 0..7
    const int m16  = lane & 15;
    const int half = lane >> 4;                 // 0 or 1
    const int aK   = half * 8;                  // A-frag K offset per lane half
    const int bK   = half * 8;                  // B-frag K offset per lane half

    const int r0 = blockIdx.x * 128 + (wave >> 1) * 32;   // 32 rows / wave
    const int c0 = blockIdx.y * 128 + (wave & 1) * 64;    // 64 cols / wave

    // A strips: rows r0+m16 and r0+16+m16
    const unsigned short* aRow0 = n1 + (size_t)(r0 + m16) * DDIM;
    const unsigned short* aRow1 = aRow0 + (size_t)16 * DDIM;
    // B strips: columns c0 + 16*cf + m16
    const unsigned short* bRow0 = n2 + (size_t)(c0 + m16) * DDIM;
    const unsigned short* bRow1 = bRow0 + (size_t)16 * DDIM;
    const unsigned short* bRow2 = bRow0 + (size_t)32 * DDIM;
    const unsigned short* bRow3 = bRow0 + (size_t)48 * DDIM;

    v8f acc[2][4] = {};

    #pragma unroll 1
    for (int k0 = 0; k0 < DDIM; k0 += 32) {
        // A fragments: elems 0..7 = K[k0+aK..], 8..15 = K[k0+16+aK..]
        v8bf a0lo = *(const v8bf*)(aRow0 + k0 + aK);
        v8bf a0hi = *(const v8bf*)(aRow0 + k0 + 16 + aK);
        v8bf a1lo = *(const v8bf*)(aRow1 + k0 + aK);
        v8bf a1hi = *(const v8bf*)(aRow1 + k0 + 16 + aK);
        // B fragments: elems 0..7 = K[k0+bK..], 8..15 = K[k0+16+bK..]
        v8bf b0lo = *(const v8bf*)(bRow0 + k0 + bK);
        v8bf b0hi = *(const v8bf*)(bRow0 + k0 + bK + 16);
        v8bf b1lo = *(const v8bf*)(bRow1 + k0 + bK);
        v8bf b1hi = *(const v8bf*)(bRow1 + k0 + bK + 16);
        v8bf b2lo = *(const v8bf*)(bRow2 + k0 + bK);
        v8bf b2hi = *(const v8bf*)(bRow2 + k0 + bK + 16);
        v8bf b3lo = *(const v8bf*)(bRow3 + k0 + bK);
        v8bf b3hi = *(const v8bf*)(bRow3 + k0 + bK + 16);

        v16bf a[2], b[4];
        a[0] = __builtin_shufflevector(a0lo, a0hi, 0,1,2,3,4,5,6,7,8,9,10,11,12,13,14,15);
        a[1] = __builtin_shufflevector(a1lo, a1hi, 0,1,2,3,4,5,6,7,8,9,10,11,12,13,14,15);
        b[0] = __builtin_shufflevector(b0lo, b0hi, 0,1,2,3,4,5,6,7,8,9,10,11,12,13,14,15);
        b[1] = __builtin_shufflevector(b1lo, b1hi, 0,1,2,3,4,5,6,7,8,9,10,11,12,13,14,15);
        b[2] = __builtin_shufflevector(b2lo, b2hi, 0,1,2,3,4,5,6,7,8,9,10,11,12,13,14,15);
        b[3] = __builtin_shufflevector(b3lo, b3hi, 0,1,2,3,4,5,6,7,8,9,10,11,12,13,14,15);

        #pragma unroll
        for (int rf = 0; rf < 2; ++rf)
            #pragma unroll
            for (int cf = 0; cf < 4; ++cf)
                acc[rf][cf] = __builtin_amdgcn_wmma_f32_16x16x32_bf16(
                        false, a[rf], false, b[cf],
                        (short)0, acc[rf][cf], false, false);
    }

    // Epilogue: acc[rf][cf][g] = sim[r0 + 16*rf + g + 8*half][c0 + 16*cf + m16].
    // Sum exp over the 4 column fragments per lane first, then one 16-lane
    // swizzle reduction and one atomic per row.
    #pragma unroll
    for (int rf = 0; rf < 2; ++rf) {
        #pragma unroll
        for (int g = 0; g < 8; ++g) {
            float e = __expf(acc[rf][0][g] * INV_T)
                    + __expf(acc[rf][1][g] * INV_T)
                    + __expf(acc[rf][2][g] * INV_T)
                    + __expf(acc[rf][3][g] * INV_T);
            e = halfwave16_sum(e);
            if (m16 == 0) {
                int row = r0 + 16 * rf + g + 8 * half;
                atomicAdd(&denom[row], e);
            }
        }
    }
}

// ---------------------------------------------------------------------------
// Kernel 5: loss = -mean(diag[i] - log(denom[i])).  Single 256-thread block.
// ---------------------------------------------------------------------------
__global__ void k_final_loss(const float* __restrict__ diag,
                             const float* __restrict__ denom,
                             float* __restrict__ out) {
    const int lane = threadIdx.x & 31;
    const int wave = threadIdx.x >> 5;

    float acc = 0.0f;
    for (int i = threadIdx.x; i < NROWS; i += blockDim.x)
        acc += diag[i] - logf(denom[i]);

    #pragma unroll
    for (int m = 1; m < 32; m <<= 1) acc += __shfl_xor(acc, m, 32);

    __shared__ float red[8];
    if (lane == 0) red[wave] = acc;
    __syncthreads();
    if (wave == 0) {
        float v = (lane < 8) ? red[lane] : 0.0f;
        #pragma unroll
        for (int m = 1; m < 8; m <<= 1) v += __shfl_xor(v, m, 32);
        if (lane == 0) out[0] = -v / (float)NROWS;
    }
}

// ---------------------------------------------------------------------------
extern "C" void kernel_launch(void* const* d_in, const int* in_sizes, int n_in,
                              void* d_out, int out_size, void* d_ws, size_t ws_size,
                              hipStream_t stream) {
    (void)in_sizes; (void)n_in; (void)out_size; (void)ws_size;

    const float* z1 = (const float*)d_in[0];
    const float* z2 = (const float*)d_in[1];
    float* out = (float*)d_out;

    char* ws = (char*)d_ws;
    const size_t mat_bytes = (size_t)NROWS * DDIM * sizeof(unsigned short); // 16 MB
    unsigned short* n1 = (unsigned short*)(ws);
    unsigned short* n2 = (unsigned short*)(ws + mat_bytes);
    float* denom = (float*)(ws + 2 * mat_bytes);
    float* diag  = denom + NROWS;

    k_normalize_bf16<<<NROWS, 256, 0, stream>>>(z1, n1);
    k_normalize_bf16<<<NROWS, 256, 0, stream>>>(z2, n2);
    k_zero_f32<<<NROWS / 256, 256, 0, stream>>>(denom, NROWS);
    k_diag_sim<<<NROWS / 8, 256, 0, stream>>>(n1, n2, diag);

    dim3 grid(NROWS / 128, NROWS / 128);   // 64 x 64 workgroups, 8 waves each
    k_gemm_exp_rowsum<<<grid, 256, 0, stream>>>(n1, n2, denom);

    k_final_loss<<<1, 256, 0, stream>>>(diag, denom, out);
}
